// DCNv2CrossNet_65429531787423
// MI455X (gfx1250) — matrix operand order repaired and checked by reading, not compiled, not run on hardware
//
#include <hip/hip_runtime.h>
#include <hip/hip_bf16.h>
#include <math.h>

// ---------------------------------------------------------------------------
// DCNv2 CrossNet for MI455X (gfx1250, wave32, WMMA).
//   B=32768, D=512, E=4 experts, low-rank=64, 3 cross layers.
// One workgroup = 16 batch rows, 4 waves = 1 wave per expert.
// bf16 WMMA (16x16x32) with f32 accumulation; whole layer chain stays in LDS.
// d_ws holds weights re-packed into the WMMA B-operand register layout
// (~1.62 MB) by a pre-pass kernel.
// x tile is brought in with GLOBAL_LOAD_ASYNC_TO_LDS_B128 (ASYNCcnt) when the
// toolchain exposes the gfx1250 async builtins.
// ---------------------------------------------------------------------------

typedef __attribute__((ext_vector_type(16))) __bf16 v16bf;
typedef __attribute__((ext_vector_type(8)))  float  v8f;

struct __align__(16) U4 { unsigned x, y, z, w; };
struct __align__(16) F4 { float x, y, z, w; };

union BF16Tile { v16bf v; U4 q[2]; };

// ---- workspace layout (dword units). One B tile = 32 lanes x 8 dwords = 256 dw.
#define TILE_DW   256
#define V_OFF     0
#define V_TILES   (12 * 16 * 4)                    // (3 layers * 4 experts) x 16 kt x 4 nt
#define C_OFF     (V_OFF + V_TILES * TILE_DW)
#define C_TILES   (12 * 2 * 4)                     // ie x 2 kt x 4 nt
#define U_OFF     (C_OFF + C_TILES * TILE_DW)
#define U_TILES   (12 * 2 * 32)                    // ie x 2 kt x 32 nt
#define G_OFF     (U_OFF + U_TILES * TILE_DW)
#define G_TILES   16                               // 16 kt (gate shared by all layers)
#define TOTAL_TILES (V_TILES + C_TILES + U_TILES + G_TILES)   // 1648

#if defined(__has_builtin)
#if __has_builtin(__builtin_amdgcn_global_load_async_to_lds_b128) && \
    __has_builtin(__builtin_amdgcn_s_wait_asynccnt)
#define HAVE_ASYNC_LDS 1
#endif
#endif

#ifdef HAVE_ASYNC_LDS
// Builtin prototype (per hipcc diagnostic): (v4i AS1*, v4i AS3*, Imm, Imm).
typedef int i4v __attribute__((vector_size(16)));
typedef __attribute__((address_space(1))) i4v gas_i4;   // global
typedef __attribute__((address_space(3))) i4v las_i4;   // LDS
// generic->AS casts via integers: LDS flat addr truncates to 32-bit LDS offset.
static __device__ inline void async_g2l_b128(const void* g, void* l) {
  __builtin_amdgcn_global_load_async_to_lds_b128(
      (gas_i4*)(unsigned long long)g,
      (las_i4*)(unsigned)(unsigned long long)l, 0, 0);
}
#endif

static __device__ inline unsigned short bfbits(float f) {
  return __builtin_bit_cast(unsigned short, (__bf16)f);
}
static __device__ inline unsigned pack_bf2(float lo, float hi) {
  return (unsigned)bfbits(lo) | ((unsigned)bfbits(hi) << 16);
}
static __device__ inline v8f zero8() {
  v8f z = {0.f, 0.f, 0.f, 0.f, 0.f, 0.f, 0.f, 0.f};
  return z;
}
static __device__ inline v8f wmma_bf16(v16bf a, v16bf b, v8f c) {
  // (neg_a, A, neg_b, B, c_mod, C, reuse_a, reuse_b)
  return __builtin_amdgcn_wmma_f32_16x16x32_bf16(false, a, false, b, (short)0, c,
                                                 false, false);
}

// B operand: contiguous K order (not ISA table order) — lanes 0..15 hold K=0..15 (VGPR r = K pair 2r,2r+1),
// lanes 16..31 hold K=16..31. Pre-packed: lane's 8 dwords are contiguous.
static __device__ inline v16bf load_b_tile(const unsigned* __restrict__ base, int lane) {
  BF16Tile t;
  const U4* p = (const U4*)base;
  t.q[0] = p[2 * lane];
  t.q[1] = p[2 * lane + 1];
  return t.v;
}

// A operand from an LDS bf16 row (row-major, M = lane&15).
// A layout matching the packed B: lane<16 -> K {0..15}, lane>=16 -> K {16..31}
// => two 16B chunks at bf16 offsets kbase+16*half and kbase+16*half+8.
static __device__ inline v16bf load_a_row(const unsigned* rowp, int kt, int half) {
  BF16Tile t;
  const U4* p = (const U4*)rowp;            // 1 U4 = 8 bf16
  t.q[0] = p[kt * 4 + 2 * half];
  t.q[1] = p[kt * 4 + 2 * half + 1];
  return t.v;
}

// ---------------------------------------------------------------------------
// Pre-pass: convert f32 weights to bf16 and pack into WMMA B-operand layout.
// One wave per 32x16 B tile. B(k,n) = W[n][k] for V/C/U (x @ Wᵀ form),
// B(k,n) = gate_w[k][n] for the gate (padded N: 4 -> 16 with zeros).
// ---------------------------------------------------------------------------
__global__ __launch_bounds__(32) void dcn_pack_weights(
    const float* __restrict__ Vs, const float* __restrict__ Cs,
    const float* __restrict__ Us, const float* __restrict__ gw,
    unsigned* __restrict__ ws)
{
  const int t    = (int)blockIdx.x;
  const int lane = (int)threadIdx.x;
  const int half = lane >> 4;
  const int nidx = lane & 15;

  const float* W = nullptr;
  int Kdim = 0, n = 0, kbase = 0;
  unsigned* outp = nullptr;
  bool gate = false;

  if (t < V_TILES) {                       // tile = ie*64 + kt*4 + nt
    int nt = t & 3, kt = (t >> 2) & 15, ie = t >> 6;
    W = Vs + (long)ie * 64 * 512; Kdim = 512;
    n = nt * 16 + nidx; kbase = kt * 32;
    outp = ws + V_OFF + (long)t * TILE_DW;
  } else if (t < V_TILES + C_TILES) {      // tile = ie*8 + kt*4 + nt
    int tc = t - V_TILES;
    int nt = tc & 3, kt = (tc >> 2) & 1, ie = tc >> 3;
    W = Cs + (long)ie * 64 * 64; Kdim = 64;
    n = nt * 16 + nidx; kbase = kt * 32;
    outp = ws + C_OFF + (long)tc * TILE_DW;
  } else if (t < V_TILES + C_TILES + U_TILES) {  // tile = ie*64 + kt*32 + nt
    int tu = t - (V_TILES + C_TILES);
    int nt = tu & 31, kt = (tu >> 5) & 1, ie = tu >> 6;
    W = Us + (long)ie * 512 * 64; Kdim = 64;
    n = nt * 16 + nidx; kbase = kt * 32;
    outp = ws + U_OFF + (long)tu * TILE_DW;
  } else {                                 // gate: tile = kt
    int kt = t - (V_TILES + C_TILES + U_TILES);
    gate = true;
    W = gw; Kdim = 512;
    n = nidx; kbase = kt * 32;
    outp = ws + G_OFF + (long)kt * TILE_DW;
  }

#pragma unroll
  for (int r = 0; r < 8; ++r) {
    int k0 = kbase + 2 * r + 16 * half;    // B layout K pair for this VGPR slot
    float lo, hi;
    if (!gate) {
      lo = W[(long)n * Kdim + k0];
      hi = W[(long)n * Kdim + k0 + 1];
    } else {
      lo = (n < 4) ? W[(long)k0 * 4 + n]       : 0.f;
      hi = (n < 4) ? W[(long)(k0 + 1) * 4 + n] : 0.f;
    }
    outp[lane * 8 + r] = pack_bf2(lo, hi);
  }
}

// ---------------------------------------------------------------------------
// Main kernel: one 16-row tile per block; 4 waves, wave e = expert e.
// ---------------------------------------------------------------------------
__global__ __launch_bounds__(128) void dcn_cross_main(
    const float* __restrict__ x, const float* __restrict__ bias,
    const unsigned* __restrict__ wsp, float* __restrict__ out)
{
  __shared__ __align__(16) float          xs0[16][512];     // x0 tile (constant)
  __shared__ __align__(16) float          xl[16][512];      // x_l, accumulated in place
  __shared__ __align__(16) unsigned       xbf[16][256];     // bf16 snapshot of x_l
  __shared__ __align__(16) unsigned short stage[4][16][64]; // per-wave tanh staging
  __shared__ __align__(16) float          biass[512];
  __shared__ float                        gstage[4][16];    // gate column per expert

  const int tid  = (int)threadIdx.x;
  const int lane = tid & 31;
  const int wave = tid >> 5;     // expert id 0..3
  const int half = lane >> 4;
  const int idx  = lane & 15;
  const long rowbase = (long)blockIdx.x * 16;

  // ---- load x tile into xs0 and xl (16 rows x 128 float4 each)
#ifdef HAVE_ASYNC_LDS
  for (int v = tid; v < 2048; v += 128) {
    int m = v >> 7, c4 = v & 127;
    const void* g = (const void*)(x + (rowbase + m) * 512 + (long)c4 * 4);
    async_g2l_b128(g, &xs0[m][c4 * 4]);
    async_g2l_b128(g, &xl[m][c4 * 4]);
  }
  __builtin_amdgcn_s_wait_asynccnt(0);
#else
  for (int v = tid; v < 2048; v += 128) {
    int m = v >> 7, c4 = v & 127;
    F4 t = ((const F4*)(x + (rowbase + m) * 512))[c4];
    ((F4*)&xs0[m][0])[c4] = t;
    ((F4*)&xl[m][0])[c4]  = t;
  }
#endif
  __syncthreads();

  for (int layer = 0; layer < 3; ++layer) {
    // stage bias and bf16 snapshot of current x_l
    ((F4*)biass)[tid & 127] = ((const F4*)(bias + layer * 512))[tid & 127];
    for (int v = tid; v < 2048; v += 128) {
      int m = v >> 7, c4 = v & 127;
      F4 t = ((const F4*)&xl[m][0])[c4];
      xbf[m][2 * c4]     = pack_bf2(t.x, t.y);
      xbf[m][2 * c4 + 1] = pack_bf2(t.z, t.w);
    }
    __syncthreads();

    const unsigned* xrow = &xbf[idx][0];   // A row: M = idx

    // ---- fused gate + V pass over K=512 (16 kt), sharing the A tiles.
    //      gate: g = x_l @ gate_w (N padded 4->16, wave keeps column `wave`)
    //      V:    v = x_l @ V[i,e]^T, N=64 (4 nt)
    // A tile is double-buffered so the ds_load latency hides under 5 WMMAs.
    const unsigned* Vb =
        wsp + V_OFF + (long)((layer * 4 + wave) * 64) * TILE_DW;
    v8f gacc = zero8();
    v8f vacc0 = zero8(), vacc1 = zero8(), vacc2 = zero8(), vacc3 = zero8();
    v16bf a = load_a_row(xrow, 0, half);
    for (int kt = 0; kt < 16; ++kt) {
      v16bf acur = a;
      if (kt < 15) a = load_a_row(xrow, kt + 1, half);
      gacc  = wmma_bf16(acur, load_b_tile(wsp + G_OFF + kt * TILE_DW, lane), gacc);
      vacc0 = wmma_bf16(acur, load_b_tile(Vb + (kt * 4 + 0) * TILE_DW, lane), vacc0);
      vacc1 = wmma_bf16(acur, load_b_tile(Vb + (kt * 4 + 1) * TILE_DW, lane), vacc1);
      vacc2 = wmma_bf16(acur, load_b_tile(Vb + (kt * 4 + 2) * TILE_DW, lane), vacc2);
      vacc3 = wmma_bf16(acur, load_b_tile(Vb + (kt * 4 + 3) * TILE_DW, lane), vacc3);
    }
    if (idx == wave) {
#pragma unroll
      for (int r = 0; r < 8; ++r) gstage[wave][r + 8 * half] = gacc[r];
    }
#pragma unroll
    for (int nt = 0; nt < 4; ++nt) {
      v8f acc = (nt == 0) ? vacc0 : (nt == 1) ? vacc1 : (nt == 2) ? vacc2 : vacc3;
#pragma unroll
      for (int r = 0; r < 8; ++r)
        stage[wave][r + 8 * half][nt * 16 + idx] = bfbits(tanhf(acc[r]));
    }

    // ---- c = tanh(v @ C[i,e]^T)  : [16,64], K=64 (2 kt), N=64 (4 nt)
    const unsigned* srow = (const unsigned*)&stage[wave][idx][0];
    v16bf va0 = load_a_row(srow, 0, half);
    v16bf va1 = load_a_row(srow, 1, half);
    const unsigned* Cb =
        wsp + C_OFF + (long)((layer * 4 + wave) * 8) * TILE_DW;
    v8f cacc0 = zero8(), cacc1 = zero8(), cacc2 = zero8(), cacc3 = zero8();
    cacc0 = wmma_bf16(va0, load_b_tile(Cb + (0 * 4 + 0) * TILE_DW, lane), cacc0);
    cacc0 = wmma_bf16(va1, load_b_tile(Cb + (1 * 4 + 0) * TILE_DW, lane), cacc0);
    cacc1 = wmma_bf16(va0, load_b_tile(Cb + (0 * 4 + 1) * TILE_DW, lane), cacc1);
    cacc1 = wmma_bf16(va1, load_b_tile(Cb + (1 * 4 + 1) * TILE_DW, lane), cacc1);
    cacc2 = wmma_bf16(va0, load_b_tile(Cb + (0 * 4 + 2) * TILE_DW, lane), cacc2);
    cacc2 = wmma_bf16(va1, load_b_tile(Cb + (1 * 4 + 2) * TILE_DW, lane), cacc2);
    cacc3 = wmma_bf16(va0, load_b_tile(Cb + (0 * 4 + 3) * TILE_DW, lane), cacc3);
    cacc3 = wmma_bf16(va1, load_b_tile(Cb + (1 * 4 + 3) * TILE_DW, lane), cacc3);
#pragma unroll
    for (int nt = 0; nt < 4; ++nt) {
      v8f acc = (nt == 0) ? cacc0 : (nt == 1) ? cacc1 : (nt == 2) ? cacc2 : cacc3;
#pragma unroll
      for (int r = 0; r < 8; ++r)
        stage[wave][r + 8 * half][nt * 16 + idx] = bfbits(tanhf(acc[r]));
    }

    // ---- u = c @ U[i,e]^T : [16,512], K=64 (2 kt), N=512 (32 nt),
    //      fused with (u + bias)*x0*g accumulated into xl via LDS f32 atomics.
    v16bf ca0 = load_a_row(srow, 0, half);
    v16bf ca1 = load_a_row(srow, 1, half);
    float gm[8];
#pragma unroll
    for (int r = 0; r < 8; ++r) gm[r] = gstage[wave][r + 8 * half];
    const unsigned* Ub =
        wsp + U_OFF + (long)((layer * 4 + wave) * 64) * TILE_DW;
    for (int nt = 0; nt < 32; ++nt) {
      v8f uacc = zero8();
      uacc = wmma_bf16(ca0, load_b_tile(Ub + (0 * 32 + nt) * TILE_DW, lane), uacc);
      uacc = wmma_bf16(ca1, load_b_tile(Ub + (1 * 32 + nt) * TILE_DW, lane), uacc);
      int n = nt * 16 + idx;
      float bn = biass[n];
#pragma unroll
      for (int r = 0; r < 8; ++r) {
        int m = r + 8 * half;
        atomicAdd(&xl[m][n], gm[r] * (uacc[r] + bn) * xs0[m][n]);  // ds_add_f32
      }
    }
    __syncthreads();   // all experts merged before next layer snapshot
  }

  // write final x_l tile
  for (int v = tid; v < 2048; v += 128) {
    int m = v >> 7, c4 = v & 127;
    ((F4*)(out + (rowbase + m) * 512))[c4] = ((const F4*)&xl[m][0])[c4];
  }
}

extern "C" void kernel_launch(void* const* d_in, const int* in_sizes, int n_in,
                              void* d_out, int out_size, void* d_ws, size_t ws_size,
                              hipStream_t stream) {
  (void)in_sizes; (void)n_in; (void)out_size; (void)ws_size;
  const float* x    = (const float*)d_in[0];
  const float* Vs   = (const float*)d_in[1];
  const float* Cs   = (const float*)d_in[2];
  const float* Us   = (const float*)d_in[3];
  const float* bias = (const float*)d_in[4];
  const float* gw   = (const float*)d_in[5];
  unsigned* ws = (unsigned*)d_ws;   // needs ~1.62 MB

  dcn_pack_weights<<<TOTAL_TILES, 32, 0, stream>>>(Vs, Cs, Us, gw, ws);
  dcn_cross_main<<<32768 / 16, 128, 0, stream>>>(x, bias, ws, (float*)d_out);
}
